// RNN_67473936220695
// MI455X (gfx1250) — matrix operand order repaired and checked by reading, not compiled, not run on hardware
//
#include <hip/hip_runtime.h>
#include <math.h>
#include <stdint.h>

typedef __bf16 bf16_t;
typedef __attribute__((ext_vector_type(16))) __bf16 v16bf;
typedef __attribute__((ext_vector_type(8)))  __bf16 v8bf;
typedef __attribute__((ext_vector_type(8)))  float  v8f;
typedef unsigned int __attribute__((ext_vector_type(4))) u32x4;
typedef int          __attribute__((ext_vector_type(8))) i32x8;
typedef int          __attribute__((ext_vector_type(4))) i32x4;

#define T_LEN 512
#define B_SZ  64
#define H_SZ  512
#define L_N   2
#define WH_RES_ROWS 256          // Wh rows pinned in LDS
#define LDS_PAD 8                // bf16 elements of row padding (bank spread)

#if defined(__has_builtin)
#  if __has_builtin(__builtin_amdgcn_tanhf)
#    define FAST_TANH(x) __builtin_amdgcn_tanhf(x)
#  elif __has_builtin(__builtin_amdgcn_tanh_f32)
#    define FAST_TANH(x) __builtin_amdgcn_tanh_f32(x)
#  endif
#  if __has_builtin(__builtin_amdgcn_tensor_load_to_lds)
#    define HAVE_TDM 1
#  endif
#endif
#ifndef FAST_TANH
#  define FAST_TANH(x) tanhf(x)
#endif

// ---------------------------------------------------------------------------
// fp32 -> bf16 weight conversion (Wi and Wh for both layers)
// ---------------------------------------------------------------------------
__global__ void convert_weights(const float* __restrict__ Wi,
                                const float* __restrict__ Wh,
                                bf16_t* __restrict__ Wibf,
                                bf16_t* __restrict__ Whbf, int n) {
  int i = blockIdx.x * blockDim.x + threadIdx.x;
  if (i < n) {
    Wibf[i] = (bf16_t)Wi[i];
    Whbf[i] = (bf16_t)Wh[i];
  }
}

// ---------------------------------------------------------------------------
// Phase 1: XW = A @ W^T + bias   (A: [M,512] fp32, W: [512,512] bf16 row=n,
// k-contiguous; output fp32 [M,512]).  One 16x16 M-tile per wave, 4 N-tiles.
// ---------------------------------------------------------------------------
__global__ __launch_bounds__(256)
void gemm_xw(const float* __restrict__ A,
             const bf16_t* __restrict__ Wbf,
             const float* __restrict__ bias,
             float* __restrict__ XW) {
  const int lane  = threadIdx.x & 31;
  const int wave  = threadIdx.x >> 5;
  const int m0    = (blockIdx.x * 8 + wave) * 16;
  const int n0    = blockIdx.y * 64;

  // CDNA5 wave32 WMMA lane mapping (ISA 7.12.2)
  const int mrow  = lane & 15;
  const int khalf = (lane < 16) ? 0 : 8;
  const int ncol  = lane & 15;
  const int koff  = (lane < 16) ? 0 : 16;
  const int mbase = (lane < 16) ? 0 : 8;

  v8f acc[4];
#pragma unroll
  for (int j = 0; j < 4; ++j) {
    float bv = bias[n0 + j * 16 + ncol];
#pragma unroll
    for (int r = 0; r < 8; ++r) acc[j][r] = bv;
  }

  const float* arow = A + (size_t)(m0 + mrow) * H_SZ;
  for (int kb = 0; kb < H_SZ; kb += 32) {
    const float* p0 = arow + kb + koff;
    const float* p1 = arow + kb + koff + 8;
    v16bf a;
#pragma unroll
    for (int j = 0; j < 8; ++j) a[j]     = (bf16_t)p0[j];
#pragma unroll
    for (int j = 0; j < 8; ++j) a[8 + j] = (bf16_t)p1[j];

#pragma unroll
    for (int j = 0; j < 4; ++j) {
      const bf16_t* bp = Wbf + (size_t)(n0 + j * 16 + ncol) * H_SZ + kb + koff;
      v16bf b = *(const v16bf*)bp;
      acc[j] = __builtin_amdgcn_wmma_f32_16x16x32_bf16(
          false, a, false, b, (short)0, acc[j], false, false);
    }
  }

#pragma unroll
  for (int j = 0; j < 4; ++j)
#pragma unroll
    for (int r = 0; r < 8; ++r)
      XW[(size_t)(m0 + mbase + r) * H_SZ + n0 + j * 16 + ncol] = acc[j][r];
}

// ---------------------------------------------------------------------------
// Phase 2: sequential scan, batch-partitioned (16 rows/block, no inter-block
// sync).  h double-buffered in LDS.  Wh rows [0,256) pinned in LDS (TDM
// preload with pad-insertion); rows [256,512) streamed from L2 each step.
// 512 threads = 16 waves.  Branchless split: every wave owns one LDS tile
// (n = wave*16) and one streamed tile (n = 256 + wave*16), so DS and VMEM
// traffic overlap per K-step and the K-loop unrolls cleanly.
// ---------------------------------------------------------------------------
__global__ __launch_bounds__(512)
void rnn_scan(const float* __restrict__ XW,    // [T*B, H]  (= x@Wi^T + bi)
              const bf16_t* __restrict__ Whbf, // [H, H] row n, k-contiguous
              const float* __restrict__ bh,    // [H]
              const float* __restrict__ h0,    // [B, H] this layer
              float* __restrict__ out,         // [T*B, H]
              float* __restrict__ hn) {        // [B, H] this layer
  __shared__ alignas(16) bf16_t whlds[WH_RES_ROWS][H_SZ + LDS_PAD]; // 266240 B
  __shared__ alignas(16) bf16_t hbuf[2][16][H_SZ + LDS_PAD];        //  33280 B

  const int lane  = threadIdx.x & 31;
  const int wave  = threadIdx.x >> 5;
  const int b0    = blockIdx.x * 16;

  const int mrow  = lane & 15;
  const int khalf = (lane < 16) ? 0 : 8;
  const int ncol  = lane & 15;
  const int koff  = (lane < 16) ? 0 : 16;
  const int mbase = (lane < 16) ? 0 : 8;

  // per-wave column bases: tile0 LDS-resident, tile1 streamed
  const int nA = wave * 16 + ncol;                 // in [0,256)
  const int nB = WH_RES_ROWS + wave * 16 + ncol;   // in [256,512)

  // ---- preload resident Wh half into LDS -------------------------------
#ifdef HAVE_TDM
  if (wave == 0) {
    // Tensor DMA descriptor: 2D tile 256 rows x 512 bf16, with LDS row
    // padding inserted by TDM (pad every 256 dwords by 4 dwords).
    uint32_t lds_base = (uint32_t)(uintptr_t)&whlds[0][0];
    uint64_t gaddr    = (uint64_t)(uintptr_t)Whbf;
    u32x4 g0;
    g0[0] = 1u;                                        // count=1, user mode
    g0[1] = lds_base;                                  // lds_addr
    g0[2] = (uint32_t)gaddr;                           // global_addr[31:0]
    g0[3] = (uint32_t)(gaddr >> 32) | (2u << 30);      // addr[56:32], type=2
    i32x8 g1;
    g1[0] = (int)((1u << 16) |                         // data_size = 2 bytes
                  (1u << 20) |                         // pad_enable
                  (7u << 22) |                         // pad_interval = 256 dw
                  (3u << 25));                         // pad_amount   = 4 dw
    g1[1] = (int)(((uint32_t)H_SZ & 0xFFFFu) << 16);   // tensor_dim0 lo16
    g1[2] = (int)(((uint32_t)WH_RES_ROWS) << 16);      // dim0 hi16=0, dim1 lo16
    g1[3] = (int)(((uint32_t)H_SZ) << 16);             // dim1 hi16=0, tile_dim0
    g1[4] = (int)WH_RES_ROWS;                          // tile_dim1 (tile_dim2=0)
    g1[5] = (int)H_SZ;                                 // tensor_dim0_stride lo32
    g1[6] = 0;                                         // stride hi16, dim1_stride lo
    g1[7] = 0;
    i32x4 gz4 = {0, 0, 0, 0};                          // 2D: groups 2/3 unused
    i32x8 gz8 = {0, 0, 0, 0, 0, 0, 0, 0};
    __builtin_amdgcn_tensor_load_to_lds(g0, g1, gz4, gz4, gz8, 0);
    __builtin_amdgcn_s_wait_tensorcnt(0);
  }
#else
  for (int idx = threadIdx.x * 8; idx < WH_RES_ROWS * H_SZ; idx += 512 * 8) {
    int r = idx >> 9, c = idx & (H_SZ - 1);
    *(v8bf*)&whlds[r][c] = *(const v8bf*)(Whbf + (size_t)r * H_SZ + c);
  }
#endif

  // ---- init h from h0 ---------------------------------------------------
  for (int idx = threadIdx.x; idx < 16 * H_SZ; idx += 512) {
    int r = idx >> 9, c = idx & (H_SZ - 1);
    hbuf[0][r][c] = (bf16_t)h0[(size_t)(b0 + r) * H_SZ + c];
  }
  __syncthreads();

  const bf16_t* wstream = Whbf + (size_t)nB * H_SZ + koff;  // streamed row base

  int cur = 0;
  for (int t = 0; t < T_LEN; ++t) {
    v8f accA, accB;
    {
      const float* xwrow = XW + (size_t)(t * B_SZ + b0 + mbase) * H_SZ;
      float bhA = bh[nA], bhB = bh[nB];
#pragma unroll
      for (int r = 0; r < 8; ++r) {
        accA[r] = xwrow[(size_t)r * H_SZ + nA] + bhA;
        accB[r] = xwrow[(size_t)r * H_SZ + nB] + bhB;
      }
    }

#pragma unroll 2
    for (int kb = 0; kb < H_SZ; kb += 32) {
      // A fragment from LDS h (two 16B ds loads per lane)
      v8bf alo = *(const v8bf*)&hbuf[cur][mrow][kb + koff];
      v8bf ahi = *(const v8bf*)&hbuf[cur][mrow][kb + koff + 8];
      v16bf a;
#pragma unroll
      for (int j = 0; j < 8; ++j) { a[j] = alo[j]; a[8 + j] = ahi[j]; }

      // tile 0: B from LDS-resident Wh rows
      v16bf bA = *(const v16bf*)&whlds[nA][kb + koff];
      // tile 1: B streamed from L2-resident Wh rows
      v16bf bB = *(const v16bf*)(wstream + kb);

      accA = __builtin_amdgcn_wmma_f32_16x16x32_bf16(
          false, a, false, bA, (short)0, accA, false, false);
      accB = __builtin_amdgcn_wmma_f32_16x16x32_bf16(
          false, a, false, bB, (short)0, accB, false, false);
    }

    int nxt = cur ^ 1;
    {
      float* orow = out + (size_t)(t * B_SZ + b0 + mbase) * H_SZ;
#pragma unroll
      for (int r = 0; r < 8; ++r) {
        float yA = FAST_TANH(accA[r]);
        float yB = FAST_TANH(accB[r]);
        orow[(size_t)r * H_SZ + nA] = yA;
        orow[(size_t)r * H_SZ + nB] = yB;
        hbuf[nxt][mbase + r][nA] = (bf16_t)yA;
        hbuf[nxt][mbase + r][nB] = (bf16_t)yB;
        if (t == T_LEN - 1) {
          hn[(size_t)(b0 + mbase + r) * H_SZ + nA] = yA;
          hn[(size_t)(b0 + mbase + r) * H_SZ + nB] = yB;
        }
      }
    }
    __syncthreads();
    cur = nxt;
  }
}

// ---------------------------------------------------------------------------
extern "C" void kernel_launch(void* const* d_in, const int* in_sizes, int n_in,
                              void* d_out, int out_size, void* d_ws, size_t ws_size,
                              hipStream_t stream) {
  const float* x    = (const float*)d_in[0];
  const float* h0   = (const float*)d_in[1];
  const float* W_ih = (const float*)d_in[2];
  const float* W_hh = (const float*)d_in[3];
  const float* b_ih = (const float*)d_in[4];
  const float* b_hh = (const float*)d_in[5];

  float* out = (float*)d_out;
  float* hn  = out + (size_t)T_LEN * B_SZ * H_SZ;

  char*   ws   = (char*)d_ws;
  float*  XW   = (float*)ws;
  bf16_t* Wibf = (bf16_t*)(ws + (size_t)T_LEN * B_SZ * H_SZ * sizeof(float));
  bf16_t* Whbf = Wibf + (size_t)L_N * H_SZ * H_SZ;

  const int nw = L_N * H_SZ * H_SZ;
  convert_weights<<<(nw + 255) / 256, 256, 0, stream>>>(W_ih, W_hh, Wibf, Whbf, nw);

  // ----- layer 0 -----
  gemm_xw<<<dim3((T_LEN * B_SZ) / (16 * 8), 8), 256, 0, stream>>>(x, Wibf, b_ih, XW);
  rnn_scan<<<B_SZ / 16, 512, 0, stream>>>(XW, Whbf, b_hh, h0, out, hn);

  // ----- layer 1 (input = layer-0 output currently living in d_out) -----
  gemm_xw<<<dim3((T_LEN * B_SZ) / (16 * 8), 8), 256, 0, stream>>>(
      out, Wibf + (size_t)H_SZ * H_SZ, b_ih + H_SZ, XW);
  rnn_scan<<<B_SZ / 16, 512, 0, stream>>>(
      XW, Whbf + (size_t)H_SZ * H_SZ, b_hh + H_SZ,
      h0 + (size_t)B_SZ * H_SZ, out, hn + (size_t)B_SZ * H_SZ);
}
